// FlashMoeBlockWrapper_12335146074516
// MI455X (gfx1250) — matrix operand order repaired and checked by reading, not compiled, not run on hardware
//
#include <hip/hip_runtime.h>

#define H_DIM 2048
#define I_DIM 768
#define I2_DIM 1536
#define E_DIM 32
#define TOPK 8
#define T_DIM 1024

typedef __attribute__((ext_vector_type(16))) __bf16 v16bf;
typedef __attribute__((ext_vector_type(8)))  float  v8f;

union BfPack {
    unsigned short s[16];
    uint4          u[2];
    v16bf          v;
};

// A-fragment (16x32 bf16): lane<16 holds K 0..15 of row M=lane;
// lane>=16 holds K 16..31 of row M=lane-16. base = row start + kk (bf16*).
__device__ __forceinline__ v16bf load_a_frag(const unsigned short* base, int lhalf) {
    BfPack p;
    const uint4* q = (const uint4*)base;   // 8 bf16 per uint4
    p.u[0] = q[2 * lhalf];
    p.u[1] = q[2 * lhalf + 1];
    return p.v;
}

// B-fragment from pre-converted bf16 weights: 16 consecutive K values per lane.
__device__ __forceinline__ v16bf load_b_frag_bf(const unsigned short* base) {
    BfPack p;
    const uint4* q = (const uint4*)base;
    p.u[0] = q[0];
    p.u[1] = q[1];
    return p.v;
}

// B-fragment from f32 weights: 16 consecutive f32 -> bf16 via v_cvt_pk_bf16_f32.
__device__ __forceinline__ v16bf cvt_b_frag(const float* base) {
    const float4* q = (const float4*)base;
    v16bf r;
#pragma unroll
    for (int i = 0; i < 4; ++i) {
        float4 f = q[i];
        r[4 * i + 0] = (__bf16)f.x;
        r[4 * i + 1] = (__bf16)f.y;
        r[4 * i + 2] = (__bf16)f.z;
        r[4 * i + 3] = (__bf16)f.w;
    }
    return r;
}

// silu(g) = g * sigmoid(g), with native v_exp_f32 + v_rcp_f32.
__device__ __forceinline__ float silu_fast(float g) {
    return g * __builtin_amdgcn_rcpf(1.0f + __expf(-g));
}

// ---------------- Kernel 1a/1b: f32 -> bf16 bulk conversion ----------------
__global__ void cvt_f32_bf16_kernel(const float* __restrict__ src, unsigned short* __restrict__ dst) {
    size_t i = (size_t)blockIdx.x * blockDim.x + threadIdx.x;   // float4 index
    float4 f = ((const float4*)src)[i];
    union { __bf16 b[4]; ushort4 u; } o;
    o.b[0] = (__bf16)f.x; o.b[1] = (__bf16)f.y; o.b[2] = (__bf16)f.z; o.b[3] = (__bf16)f.w;
    ((ushort4*)dst)[i] = o.u;
}

// ---------------- Kernel 2: gating (softmax + top-8 + renorm) ----------------
// One wave32 per token; lane == expert (E == 32 == wave width).
__global__ void gate_kernel(const float* __restrict__ x, const float* __restrict__ gw,
                            float* __restrict__ combine) {
    const int t = blockIdx.x;
    const int lane = threadIdx.x;

    const float4* xr = (const float4*)(x + (size_t)t * H_DIM);
    const float4* gr = (const float4*)(gw + (size_t)lane * H_DIM);
    float acc = 0.0f;
    for (int i = 0; i < H_DIM / 4; ++i) {
        float4 a = xr[i], b = gr[i];
        acc += a.x * b.x + a.y * b.y + a.z * b.z + a.w * b.w;
    }
    // softmax numerator (top-k renormalization cancels the denominator)
    float m = acc;
    for (int off = 16; off >= 1; off >>= 1) m = fmaxf(m, __shfl_xor(m, off));
    float z = __expf(acc - m);

    float p = z, myw = 0.0f, wsum = 0.0f;
    int sel = 0;
#pragma unroll
    for (int i = 0; i < TOPK; ++i) {
        float mx = p;
        for (int off = 16; off >= 1; off >>= 1) mx = fmaxf(mx, __shfl_xor(mx, off));
        unsigned long long msk = __ballot(p == mx);
        int ldr = __ffsll((unsigned long long)msk) - 1;
        wsum += mx;
        if (lane == ldr) { myw = p; sel = 1; p = -1.0f; }
    }
    combine[(size_t)t * E_DIM + lane] = sel ? (myw * __builtin_amdgcn_rcpf(wsum)) : 0.0f;
}

// ---------------- Kernel 3: h = x @ w13[e]^T, act = silu(gate)*up -> bf16 ----------------
// grid: (I/32, T/128, E); block: 256 (8 waves as 4M x 2N). Each wave holds TWO
// 16-row M-subtiles sharing each B fragment (4 accumulators: gate x2, up x2).
// Block tile M=128 -> w13 is re-fetched only T/128 = 8 times total, and the
// per-expert 12.6 MB slice stays L2-resident across the token sweep.
__global__ __launch_bounds__(256) void gemm1_silu_kernel(const unsigned short* __restrict__ xb,
                                                         const float* __restrict__ w13,
                                                         unsigned short* __restrict__ actb) {
    const int e     = blockIdx.z;
    const int tBase = blockIdx.y * 128;
    const int wave  = threadIdx.x >> 5;
    const int lane  = threadIdx.x & 31;
    const int l16   = lane & 15;
    const int lhalf = lane >> 4;
    const int mw    = wave & 3;          // 4 M groups of 32 rows
    const int nw    = wave >> 2;         // 2 N groups of 16 gate cols
    const int rowT  = tBase + mw * 32;
    const int cTile = blockIdx.x * 32 + nw * 16;

    const unsigned short* xrow0 = xb + (size_t)(rowT + l16) * H_DIM;
    const unsigned short* xrow1 = xb + (size_t)(rowT + 16 + l16) * H_DIM;
    const float* wg = w13 + ((size_t)e * I2_DIM + cTile + l16) * H_DIM;          // gate row
    const float* wu = w13 + ((size_t)e * I2_DIM + cTile + l16 + I_DIM) * H_DIM;  // up row

    v8f cg0 = {}, cg1 = {}, cu0 = {}, cu1 = {};
    for (int kk = 0; kk < H_DIM; kk += 32) {
        __builtin_prefetch(wg + kk + 128, 0, 1);
        __builtin_prefetch(wu + kk + 128, 0, 1);
        v16bf a0 = load_a_frag(xrow0 + kk, lhalf);
        v16bf a1 = load_a_frag(xrow1 + kk, lhalf);
        v16bf bg = cvt_b_frag(wg + kk + lhalf * 16);
        v16bf bu = cvt_b_frag(wu + kk + lhalf * 16);
        cg0 = __builtin_amdgcn_wmma_f32_16x16x32_bf16(false, a0, false, bg, (short)0, cg0, false, false);
        cu0 = __builtin_amdgcn_wmma_f32_16x16x32_bf16(false, a0, false, bu, (short)0, cu0, false, false);
        cg1 = __builtin_amdgcn_wmma_f32_16x16x32_bf16(false, a1, false, bg, (short)0, cg1, false, false);
        cu1 = __builtin_amdgcn_wmma_f32_16x16x32_bf16(false, a1, false, bu, (short)0, cu1, false, false);
    }
    // C layout: VGPR r -> row (r + 8*lhalf), col l16
#pragma unroll
    for (int s = 0; s < 2; ++s) {
        const v8f& cg = s ? cg1 : cg0;
        const v8f& cu = s ? cu1 : cu0;
#pragma unroll
        for (int r = 0; r < 8; ++r) {
            float a = silu_fast(cg[r]) * cu[r];
            union { __bf16 b; unsigned short us; } o;
            o.b = (__bf16)a;
            actb[((size_t)e * T_DIM + rowT + s * 16 + r + 8 * lhalf) * I_DIM + cTile + l16] = o.us;
        }
    }
}

// ---------------- Kernel 4: out = sum_e combine[t,e] * (act[e] @ w2b[e]^T) ----------------
// grid: (H/128, T/64); block: 256 (8 waves as 4M x 2N). Wave tile: 16 rows x 64 cols
// (4 accumulators). Per-expert partial scaled by combine (LDS, async-staged) before
// folding. Both operands are bf16 (148 MB combined -> L2-resident) -> inner loop is
// pure b128 loads + WMMA.
__global__ __launch_bounds__(256) void gemm2_combine_kernel(const unsigned short* __restrict__ actb,
                                                            const unsigned short* __restrict__ w2b,
                                                            const float* __restrict__ combine,
                                                            float* __restrict__ out) {
    __shared__ float comb[64 * 32];
    const int tBase = blockIdx.y * 64;
    const int hBase = blockIdx.x * 128;

    // Stage the (contiguous) 64x32 combine tile into LDS via async DMA:
    // 2048 floats = 512 x b128 chunks; 2 chunks per thread. GV addressing form.
    {
        const float* gsrc = combine + (size_t)tBase * E_DIM;
#pragma unroll
        for (int j = 0; j < 2; ++j) {
            int c = threadIdx.x + 256 * j;                     // b128 chunk index
            unsigned long long gaddr = (unsigned long long)(gsrc + c * 4);
            unsigned ldsa = (unsigned)(unsigned long long)(&comb[c * 4]);
            asm volatile("global_load_async_to_lds_b128 %0, %1, off"
                         :: "v"(ldsa), "v"(gaddr) : "memory");
        }
        asm volatile("s_wait_asynccnt 0x0" ::: "memory");
    }
    __syncthreads();

    const int wave  = threadIdx.x >> 5;
    const int lane  = threadIdx.x & 31;
    const int l16   = lane & 15;
    const int lhalf = lane >> 4;
    const int mw    = wave & 3;
    const int nw    = wave >> 2;
    const int rowT  = tBase + mw * 16;
    const int colH  = hBase + nw * 64;

    v8f oacc[4] = {{}, {}, {}, {}};
    for (int e = 0; e < E_DIM; ++e) {
        v8f acc[4] = {{}, {}, {}, {}};
        const unsigned short* arow = actb + ((size_t)e * T_DIM + rowT + l16) * I_DIM;
        const unsigned short* w2e  = w2b + (size_t)e * H_DIM * I_DIM;
        for (int kk = 0; kk < I_DIM; kk += 32) {
            __builtin_prefetch(arow + kk + 64, 0, 1);
            v16bf a = load_a_frag(arow + kk, lhalf);
#pragma unroll
            for (int n = 0; n < 4; ++n) {
                const unsigned short* wr =
                    w2e + (size_t)(colH + n * 16 + l16) * I_DIM + kk + lhalf * 16;
                v16bf b = load_b_frag_bf(wr);
                acc[n] = __builtin_amdgcn_wmma_f32_16x16x32_bf16(false, a, false, b, (short)0,
                                                                 acc[n], false, false);
            }
        }
#pragma unroll
        for (int r = 0; r < 8; ++r) {
            float cw = comb[(mw * 16 + r + 8 * lhalf) * 32 + e];
#pragma unroll
            for (int n = 0; n < 4; ++n) oacc[n][r] += cw * acc[n][r];
        }
    }
#pragma unroll
    for (int n = 0; n < 4; ++n)
#pragma unroll
        for (int r = 0; r < 8; ++r)
            out[(size_t)(rowT + r + 8 * lhalf) * H_DIM + colH + n * 16 + l16] = oacc[n][r];
}

extern "C" void kernel_launch(void* const* d_in, const int* in_sizes, int n_in,
                              void* d_out, int out_size, void* d_ws, size_t ws_size,
                              hipStream_t stream) {
    const float* x   = (const float*)d_in[0];  // (T, H)
    const float* gw  = (const float*)d_in[1];  // (E, H)
    const float* w13 = (const float*)d_in[2];  // (E, 2I, H)
    const float* w2  = (const float*)d_in[3];  // (E, H, I)
    float* out = (float*)d_out;                // (T, H)

    char* ws = (char*)d_ws;
    size_t off = 0;
    unsigned short* xb   = (unsigned short*)(ws + off); off += (size_t)T_DIM * H_DIM * 2;          // 4 MB
    unsigned short* actb = (unsigned short*)(ws + off); off += (size_t)E_DIM * T_DIM * I_DIM * 2;  // 48 MB
    unsigned short* w2b  = (unsigned short*)(ws + off); off += (size_t)E_DIM * H_DIM * I_DIM * 2;  // 100 MB
    float* combine = (float*)(ws + off);                                                           // 128 KB

    cvt_f32_bf16_kernel<<<(T_DIM * H_DIM / 4) / 256, 256, 0, stream>>>(x, xb);
    cvt_f32_bf16_kernel<<<((size_t)E_DIM * H_DIM * I_DIM / 4) / 256, 256, 0, stream>>>(w2, w2b);
    gate_kernel<<<T_DIM, 32, 0, stream>>>(x, gw, combine);
    gemm1_silu_kernel<<<dim3(I_DIM / 32, T_DIM / 128, E_DIM), 256, 0, stream>>>(xb, w13, actb);
    gemm2_combine_kernel<<<dim3(H_DIM / 128, T_DIM / 64), 256, 0, stream>>>(actb, w2b, combine, out);
}
